// LSTM_24661702213955
// MI455X (gfx1250) — matrix operand order repaired and checked by reading, not compiled, not run on hardware
//
#include <hip/hip_runtime.h>
#include <hip/hip_bf16.h>

// -------- problem constants --------
#define LSTM_B   64
#define LSTM_S   512
#define LSTM_IN  1024
#define LSTM_H   1024
#define LSTM_G   4096   // 4*H

typedef __attribute__((ext_vector_type(16))) __bf16 bf16x16;
typedef __attribute__((ext_vector_type(8)))  float  f32x8;
typedef __attribute__((ext_vector_type(4)))  unsigned int u32x4;
typedef __attribute__((ext_vector_type(8)))  int  i32x8;
typedef __attribute__((ext_vector_type(4)))  int  i32x4;

union Frag {           // one WMMA 16-bf16-per-lane operand (32 bytes)
    bf16x16 v;
    uint4   q[2];
};

#if __has_builtin(__builtin_amdgcn_tensor_load_to_lds)
#define HAVE_TDM 1
#else
#define HAVE_TDM 0
#endif

// ---------------------------------------------------------------------------
// f32 -> bf16 conversion (grid-stride)
// ---------------------------------------------------------------------------
__global__ void k_f32_to_bf16(const float* __restrict__ src,
                              unsigned short* __restrict__ dst, int n)
{
    __bf16* d = (__bf16*)dst;
    int i = blockIdx.x * blockDim.x + threadIdx.x;
    int stride = gridDim.x * blockDim.x;
    for (; i < n; i += stride) d[i] = (__bf16)src[i];
}

__global__ void k_bias_sum(const float* __restrict__ a, const float* __restrict__ b,
                           float* __restrict__ o, int n)
{
    int i = blockIdx.x * blockDim.x + threadIdx.x;
    if (i < n) o[i] = a[i] + b[i];
}

// zero h ping-pong buffers + barrier counters before each recurrence launch
__global__ void k_init_rec(unsigned short* __restrict__ hbuf, unsigned* __restrict__ cnt)
{
    int i = blockIdx.x * blockDim.x + threadIdx.x;
    int stride = gridDim.x * blockDim.x;
    const int n = 2 * LSTM_B * LSTM_H;
    for (int j = i; j < n; j += stride) hbuf[j] = 0;
    if (blockIdx.x == 0 && threadIdx.x < 4) cnt[threadIdx.x] = 0u;
}

// ---------------------------------------------------------------------------
// bf16 GEMM with f32 accumulate:  out[M,N] = A[M,K] * W[N,K]^T + bias[N]
// block = 256 threads = 8 waves; wave tile = 32(M) x 64(N); block = 64 x 256
// ---------------------------------------------------------------------------
__global__ __launch_bounds__(256)
void k_gemm_bf16(const unsigned short* __restrict__ A_,
                 const unsigned short* __restrict__ W_,
                 const float* __restrict__ bias,
                 float* __restrict__ out,
                 int M, int N, int K)
{
    const __bf16* A = (const __bf16*)A_;
    const __bf16* W = (const __bf16*)W_;

    const int lane  = threadIdx.x & 31;
    const int half  = lane >> 4;
    const int ln    = lane & 15;
    const int wid   = threadIdx.x >> 5;
    const int waveM = wid & 1;
    const int waveN = wid >> 1;
    const int tm = blockIdx.x * 64 + waveM * 32;
    const int tn = blockIdx.y * 256 + waveN * 64;

    f32x8 acc[2][4];
#pragma unroll
    for (int i = 0; i < 2; ++i)
#pragma unroll
        for (int j = 0; j < 4; ++j) {
            float bv = bias[tn + j * 16 + ln];
#pragma unroll
            for (int r = 0; r < 8; ++r) acc[i][j][r] = bv;
        }

    const __bf16* aRow0 = A + (size_t)(tm + ln) * K;
    const __bf16* aRow1 = A + (size_t)(tm + 16 + ln) * K;

#pragma unroll 2
    for (int kk = 0; kk < K; kk += 32) {
        Frag a[2];
        // ISA A-frag layout: K = half*8+[0..8) in v0-3, K = 16+half*8+[0..8) in v4-7
        a[0].q[0] = *(const uint4*)(aRow0 + kk + half * 8);
        a[0].q[1] = *(const uint4*)(aRow0 + kk + 16 + half * 8);
        a[1].q[0] = *(const uint4*)(aRow1 + kk + half * 8);
        a[1].q[1] = *(const uint4*)(aRow1 + kk + 16 + half * 8);
#pragma unroll
        for (int j = 0; j < 4; ++j) {
            // ISA B-frag layout: K = half*8+[0..8) in v0-3, K = 16+half*8+[0..8) in v4-7
            const __bf16* wRow = W + (size_t)(tn + j * 16 + ln) * K + kk + half * 8;
            Frag b;
            b.q[0] = *(const uint4*)(wRow);
            b.q[1] = *(const uint4*)(wRow + 16);
#pragma unroll
            for (int i = 0; i < 2; ++i)
                acc[i][j] = __builtin_amdgcn_wmma_f32_16x16x32_bf16(
                    false, a[i].v, false, b.v, (short)0, acc[i][j], false, false);
        }
    }

#pragma unroll
    for (int i = 0; i < 2; ++i)
#pragma unroll
        for (int j = 0; j < 4; ++j)
#pragma unroll
            for (int r = 0; r < 8; ++r) {
                int m = tm + i * 16 + r + half * 8;
                int n = tn + j * 16 + ln;
                out[(size_t)m * N + n] = acc[i][j][r];
            }
}

// ---------------------------------------------------------------------------
// TDM: DMA one per-block pre-gate tile  [16 batch][4 gates][64 j] f32 -> LDS.
//   x: 64 contiguous f32; y: 4 gates, stride H elems; z: 16 rows, stride S*4H.
// ---------------------------------------------------------------------------
#if HAVE_TDM
__device__ __forceinline__ void tdm_load_pre_tile(const float* gsrc, unsigned ldsOff)
{
    unsigned long long ga = (unsigned long long)gsrc;
    u32x4 g0;
    g0[0] = 1u;                                        // count=1, user descriptor
    g0[1] = ldsOff;                                    // lds_addr (bytes)
    g0[2] = (unsigned)ga;                              // global_addr[31:0]
    g0[3] = (unsigned)((ga >> 32) & 0x1FFFFFFull)      // global_addr[56:32]
            | (2u << 30);                              // type = 2 ("image")
    i32x8 g1;
    g1[0] = 0x00020000;                                // data_size = 2 (4 bytes)
    g1[1] = (64 << 16);                                // tensor_dim0 = 64
    g1[2] = (4 << 16);                                 // tensor_dim1 = 4
    g1[3] = (64 << 16);                                // tile_dim0 = 64
    g1[4] = 4 | (16 << 16);                            // tile_dim1 = 4, tile_dim2 = 16
    g1[5] = LSTM_H;                                    // tensor_dim0_stride = 1024
    g1[6] = 0;                                         // stride0[47:32]=0 | stride1[15:0]=0
    g1[7] = (int)((unsigned)(LSTM_S * LSTM_G) >> 16);  // stride1[47:16] (S*4H = 0x200000)
    i32x4 g2 = {16, 0, 0, 0};                          // tensor_dim2 = 16
    i32x4 g3 = {0, 0, 0, 0};
#if defined(__clang_major__) && (__clang_major__ >= 23)
    i32x8 g4 = {0, 0, 0, 0, 0, 0, 0, 0};
    __builtin_amdgcn_tensor_load_to_lds(g0, g1, g2, g3, g4, 0);
#else
    __builtin_amdgcn_tensor_load_to_lds(g0, g1, g2, g3, 0);
#endif
}
#endif

// ---------------------------------------------------------------------------
// Persistent LSTM recurrence. 64 blocks x 256 threads.
//   block -> (batch group bg in [0,4): 16 rows, hidden slice ns in [0,16): 64 j)
//   per step: gates[16 x 256] = pre-slice + h(16x1024) @ Wh_slice^T  (WMMA)
//   pre-slice for step t+1 is TDM-prefetched into LDS during step t.
//   per-batch-group atomic barrier (only 16 blocks share h rows).
// ---------------------------------------------------------------------------
__global__ __launch_bounds__(256)
void k_lstm_rec(const float* __restrict__ pre,            // [B,S,4H] f32
                const unsigned short* __restrict__ Wh_,   // [4H,H] bf16
                unsigned short* __restrict__ hbuf_,       // [2,B,H] bf16 ping-pong
                float* __restrict__ outF32,               // [B,S,H] or null
                unsigned short* __restrict__ outB16,      // [B,S,H] bf16 or null
                float* __restrict__ hn,                   // [B,H]
                float* __restrict__ cn,                   // [B,H]
                unsigned* __restrict__ cnt)               // [4] barrier counters
{
    __shared__ float gbuf[16 * 256];
#if HAVE_TDM
    __shared__ float preLds[2][16 * 4 * 64];              // double-buffered pre tile
#endif

    const __bf16* Wh = (const __bf16*)Wh_;
    __bf16* hbuf = (__bf16*)hbuf_;

    const int tid  = threadIdx.x;
    const int lane = tid & 31;
    const int half = lane >> 4;
    const int ln   = lane & 15;
    const int wid  = tid >> 5;
    const int bg   = blockIdx.x & 3;     // batch group (16 rows)
    const int ns   = blockIdx.x >> 2;    // hidden slice (64 j)
    const int bRow0 = bg * 16;
    const int j0    = ns * 64;

    // wave handles local gate-columns [wid*32, wid*32+32): lc -> gate=lc>>6, jj=lc&63
    int gcol[2], lgate[2], ljj[2];
    const __bf16* wRow[2];
#pragma unroll
    for (int tj = 0; tj < 2; ++tj) {
        int lc = wid * 32 + tj * 16 + ln;
        lgate[tj] = lc >> 6;
        ljj[tj]   = lc & 63;
        gcol[tj]  = lgate[tj] * LSTM_H + j0 + ljj[tj];
        wRow[tj]  = Wh + (size_t)gcol[tj] * LSTM_H;
    }

    float cst[4];
#pragma unroll
    for (int e = 0; e < 4; ++e) cst[e] = 0.0f;

#if HAVE_TDM
    const float* preBase = pre + (size_t)bRow0 * LSTM_S * LSTM_G + j0;
    if (wid == 0) {
        tdm_load_pre_tile(preBase, (unsigned)(size_t)(void*)&preLds[0][0]);
        __builtin_amdgcn_s_wait_tensorcnt(0);
    }
    __syncthreads();
#endif

    for (int t = 0; t < LSTM_S; ++t) {
        const __bf16* hRd = hbuf + (size_t)(t & 1) * LSTM_B * LSTM_H;
        __bf16*       hWr = hbuf + (size_t)((t + 1) & 1) * LSTM_B * LSTM_H;

#if HAVE_TDM
        // prefetch next step's pre tile while this step computes
        if (wid == 0 && (t + 1) < LSTM_S)
            tdm_load_pre_tile(preBase + (size_t)(t + 1) * LSTM_G,
                              (unsigned)(size_t)(void*)&preLds[(t + 1) & 1][0]);
#endif

        // accumulators start at precomputed input gates
        f32x8 acc[2];
#if HAVE_TDM
        const float* pl = &preLds[t & 1][0];
#pragma unroll
        for (int tj = 0; tj < 2; ++tj)
#pragma unroll
            for (int r = 0; r < 8; ++r)
                acc[tj][r] = pl[((r + half * 8) * 4 + lgate[tj]) * 64 + ljj[tj]];
#else
#pragma unroll
        for (int tj = 0; tj < 2; ++tj)
#pragma unroll
            for (int r = 0; r < 8; ++r) {
                int b = bRow0 + r + half * 8;
                acc[tj][r] = pre[((size_t)b * LSTM_S + t) * LSTM_G + gcol[tj]];
            }
#endif

        const __bf16* hRow = hRd + (size_t)(bRow0 + ln) * LSTM_H;
#pragma unroll 2
        for (int kk = 0; kk < LSTM_H; kk += 32) {
            Frag a;
            a.q[0] = *(const uint4*)(hRow + kk + half * 8);
            a.q[1] = *(const uint4*)(hRow + kk + 16 + half * 8);
#pragma unroll
            for (int tj = 0; tj < 2; ++tj) {
                Frag b;
                b.q[0] = *(const uint4*)(wRow[tj] + kk + half * 8);
                b.q[1] = *(const uint4*)(wRow[tj] + kk + 16 + half * 8);
                acc[tj] = __builtin_amdgcn_wmma_f32_16x16x32_bf16(
                    false, a.v, false, b.v, (short)0, acc[tj], false, false);
            }
        }

        // stage gates in LDS for the elementwise phase
#pragma unroll
        for (int tj = 0; tj < 2; ++tj) {
            int lc = wid * 32 + tj * 16 + ln;
#pragma unroll
            for (int r = 0; r < 8; ++r)
                gbuf[(r + half * 8) * 256 + lc] = acc[tj][r];
        }
        __syncthreads();

        // elementwise gates: 256 threads x 4 elements cover 16 x 64 (m, jj)
#pragma unroll
        for (int e = 0; e < 4; ++e) {
            int idx = tid * 4 + e;
            int m  = idx >> 6;
            int jj = idx & 63;
            float iv = gbuf[m * 256 + 0 * 64 + jj];
            float fv = gbuf[m * 256 + 1 * 64 + jj];
            float gv = gbuf[m * 256 + 2 * 64 + jj];
            float ov = gbuf[m * 256 + 3 * 64 + jj];
            iv = 1.0f / (1.0f + __expf(-iv));
            fv = 1.0f / (1.0f + __expf(-fv));
            gv = tanhf(gv);
            ov = 1.0f / (1.0f + __expf(-ov));
            float cNew = fv * cst[e] + iv * gv;
            cst[e] = cNew;
            float hNew = ov * tanhf(cNew);

            int b = bRow0 + m;
            int j = j0 + jj;
            hWr[(size_t)b * LSTM_H + j] = (__bf16)hNew;
            if (outF32) outF32[((size_t)b * LSTM_S + t) * LSTM_H + j] = hNew;
            if (outB16) ((__bf16*)outB16)[((size_t)b * LSTM_S + t) * LSTM_H + j] = (__bf16)hNew;
            if (t == LSTM_S - 1) {
                hn[(size_t)b * LSTM_H + j] = hNew;
                cn[(size_t)b * LSTM_H + j] = cNew;
            }
        }
#if HAVE_TDM
        if (wid == 0) __builtin_amdgcn_s_wait_tensorcnt(0);   // t+1 tile landed
#endif
        __threadfence();      // make h stores visible at agent scope
        __syncthreads();

        // barrier across the 16 blocks of this batch group (monotonic counter)
        if (tid == 0) {
            __hip_atomic_fetch_add(&cnt[bg], 1u, __ATOMIC_RELEASE, __HIP_MEMORY_SCOPE_AGENT);
            unsigned target = 16u * (unsigned)(t + 1);
            while (__hip_atomic_load(&cnt[bg], __ATOMIC_ACQUIRE, __HIP_MEMORY_SCOPE_AGENT) < target)
                __builtin_amdgcn_s_sleep(2);
        }
        __syncthreads();
        __builtin_amdgcn_fence(__ATOMIC_ACQUIRE, "agent");  // invalidate L0 before re-reading h
    }
}

// ---------------------------------------------------------------------------
// Host-side orchestration
// ---------------------------------------------------------------------------
extern "C" void kernel_launch(void* const* d_in, const int* in_sizes, int n_in,
                              void* d_out, int out_size, void* d_ws, size_t ws_size,
                              hipStream_t stream)
{
    const float* x   = (const float*)d_in[0];
    const float* Wi0 = (const float*)d_in[1];
    const float* Wh0 = (const float*)d_in[2];
    const float* bi0 = (const float*)d_in[3];
    const float* bh0 = (const float*)d_in[4];
    const float* Wi1 = (const float*)d_in[5];
    const float* Wh1 = (const float*)d_in[6];
    const float* bi1 = (const float*)d_in[7];
    const float* bh1 = (const float*)d_in[8];

    char* ws = (char*)d_ws;
    // workspace layout (~638 MB total)
    float*          pre   = (float*)(ws);                              // 512 MB
    unsigned short* abuf  = (unsigned short*)(ws + 536870912ull);      //  64 MB (bf16 layer input)
    unsigned short* wi0b  = (unsigned short*)(ws + 603979776ull);      //   8 MB
    unsigned short* wh0b  = (unsigned short*)(ws + 612368384ull);      //   8 MB
    unsigned short* wi1b  = (unsigned short*)(ws + 620756992ull);      //   8 MB
    unsigned short* wh1b  = (unsigned short*)(ws + 629145600ull);      //   8 MB
    float*          bias0 = (float*)(ws + 637534208ull);               //  16 KB
    float*          bias1 = (float*)(ws + 637550592ull);               //  16 KB
    unsigned short* hbuf  = (unsigned short*)(ws + 637566976ull);      // 256 KB
    unsigned*       cnt   = (unsigned*)(ws + 637829120ull);            //  16 B

    float* outSeq = (float*)d_out;                                     // [B,S,H]
    float* hnBase = outSeq + (size_t)LSTM_B * LSTM_S * LSTM_H;         // [2,B,H]
    float* cnBase = hnBase + 2ull * LSTM_B * LSTM_H;                   // [2,B,H]

    // one-time conversions (L2-resident working set afterwards)
    k_f32_to_bf16<<<2048, 256, 0, stream>>>(x,   abuf, LSTM_B * LSTM_S * LSTM_IN);
    k_f32_to_bf16<<<1024, 256, 0, stream>>>(Wi0, wi0b, LSTM_G * LSTM_IN);
    k_f32_to_bf16<<<1024, 256, 0, stream>>>(Wh0, wh0b, LSTM_G * LSTM_H);
    k_f32_to_bf16<<<1024, 256, 0, stream>>>(Wi1, wi1b, LSTM_G * LSTM_H);
    k_f32_to_bf16<<<1024, 256, 0, stream>>>(Wh1, wh1b, LSTM_G * LSTM_H);
    k_bias_sum<<<16, 256, 0, stream>>>(bi0, bh0, bias0, LSTM_G);
    k_bias_sum<<<16, 256, 0, stream>>>(bi1, bh1, bias1, LSTM_G);

    dim3 gg(LSTM_B * LSTM_S / 64, LSTM_G / 256);

    // ---- layer 0 ----
    k_gemm_bf16<<<gg, 256, 0, stream>>>(abuf, wi0b, bias0, pre,
                                        LSTM_B * LSTM_S, LSTM_G, LSTM_IN);
    k_init_rec<<<256, 256, 0, stream>>>(hbuf, cnt);
    k_lstm_rec<<<64, 256, 0, stream>>>(pre, wh0b, hbuf,
                                       nullptr, abuf,           // bf16 seq -> layer1 input
                                       hnBase, cnBase, cnt);

    // ---- layer 1 ----
    k_gemm_bf16<<<gg, 256, 0, stream>>>(abuf, wi1b, bias1, pre,
                                        LSTM_B * LSTM_S, LSTM_G, LSTM_H);
    k_init_rec<<<256, 256, 0, stream>>>(hbuf, cnt);
    k_lstm_rec<<<64, 256, 0, stream>>>(pre, wh1b, hbuf,
                                       outSeq, nullptr,
                                       hnBase + (size_t)LSTM_B * LSTM_H,
                                       cnBase + (size_t)LSTM_B * LSTM_H, cnt);
}
